// A3TGCN_Temporal_16561393893836
// MI455X (gfx1250) — matrix operand order repaired and checked by reading, not compiled, not run on hardware
//
#include <hip/hip_runtime.h>

typedef __attribute__((ext_vector_type(16))) _Float16 v16h;
typedef __attribute__((ext_vector_type(8)))  float    v8f;

#define FDIM 8
#define TDIM 12
#define ODIM 24
#define C96  (FDIM*TDIM)

// ---------------- prep: softmax(att), fused matrices Mz/Mh, fused biases cz/ch ---------
// prep layout (floats): [0..11] probs, [16..39] cz, [48..71] ch, [80..271] Mz, [272..463] Mh
__global__ void prep_kernel(const float* __restrict__ att,
                            const float* __restrict__ Wcz, const float* __restrict__ bcz,
                            const float* __restrict__ Wch, const float* __restrict__ bch,
                            const float* __restrict__ Wlz, const float* __restrict__ blz,
                            const float* __restrict__ Wlh, const float* __restrict__ blh,
                            float* __restrict__ prep)
{
    int lane = threadIdx.x;
    if (lane < TDIM) {                               // softmax over periods
        float m = -1e30f;
        for (int i = 0; i < TDIM; ++i) m = fmaxf(m, att[i]);
        float s = 0.f;
        for (int i = 0; i < TDIM; ++i) s += __expf(att[i] - m);
        prep[lane] = __expf(att[lane] - m) / s;
    }
    if (lane < ODIM) {                               // fused biases (H==0 -> only top half of Wl*)
        float sz = blz[lane], sh = blh[lane];
        for (int j = 0; j < ODIM; ++j) {
            sz += bcz[j] * Wlz[j*ODIM + lane];
            sh += bch[j] * Wlh[j*ODIM + lane];
        }
        prep[16 + lane] = sz;
        prep[48 + lane] = sh;
    }
    for (int idx = lane; idx < FDIM*ODIM; idx += 32) {   // Mz = Wcz @ Wlz[:24], Mh = Wch @ Wlh[:24]
        int k = idx / ODIM, c = idx % ODIM;
        float sz = 0.f, sh = 0.f;
        for (int j = 0; j < ODIM; ++j) {
            sz += Wcz[k*ODIM + j] * Wlz[j*ODIM + c];
            sh += Wch[k*ODIM + j] * Wlh[j*ODIM + c];
        }
        prep[80  + idx] = sz;
        prep[272 + idx] = sh;
    }
}

// ---------------- GCN degree / norm ----------------
__global__ void deg_init(float* __restrict__ deg, int N) {
    int i = blockIdx.x * blockDim.x + threadIdx.x;
    if (i < N) deg[i] = 1.0f;                        // self-loop weight
}

__global__ void deg_edges(const int* __restrict__ ei, const float* __restrict__ w,
                          float* __restrict__ deg, int E) {
    int e = blockIdx.x * blockDim.x + threadIdx.x;
    if (e < E) atomicAdd(&deg[ei[E + e]], w[e]);     // dst-side degree
}

__global__ void deg_to_dinv(float* __restrict__ deg, int N) {
    int i = blockIdx.x * blockDim.x + threadIdx.x;
    if (i < N) {
        float d = deg[i];
        deg[i] = (d > 0.f) ? rsqrtf(d) : 0.f;
    }
}

// ---------------- SpMM: agg = A_norm @ X, agg layout [N][T][F] ----------------
__global__ void agg_init(const float* __restrict__ x, const float* __restrict__ dinv,
                         float* __restrict__ agg, int N) {
    long gid = (long)blockIdx.x * blockDim.x + threadIdx.x;
    if (gid >= (long)N * C96) return;
    int i = (int)(gid / C96), c = (int)(gid % C96);
    int t = c >> 3, f = c & 7;
    float di = dinv[i];                              // self-loop: dinv[i]^2 * x[i]
    agg[(size_t)i*C96 + c] = di * di * x[(size_t)i*C96 + f*TDIM + t];
}

__global__ void agg_scatter(const float* __restrict__ x, const int* __restrict__ ei,
                            const float* __restrict__ w, const float* __restrict__ dinv,
                            float* __restrict__ agg, int E) {
    long gid = (long)blockIdx.x * blockDim.x + threadIdx.x;
    if (gid >= (long)E * C96) return;
    int e = (int)(gid / C96), c = (int)(gid % C96);
    int s = ei[e], d = ei[E + e];
    float nrm = dinv[s] * w[e] * dinv[d];
    int t = c >> 3, f = c & 7;
    float v = nrm * x[(size_t)s*C96 + f*TDIM + t];
    atomicAdd(&agg[(size_t)d*C96 + c], v);           // L2-resident (agg = 19.2 MB << 192 MB L2)
}

// Branch-free activations: (1 - sigmoid(a)) * tanh(b).
// (1-sigmoid(a)) = 1/(1+e^a);  tanh(b) = copysign((1-t)/(1+t), b), t = e^(-2|b|) in (0,1].
// Uses hardware v_rcp_f32 (we already approximate via __expf, so IEEE divide buys nothing).
// For padded columns a=b=0 -> result is exactly 0.
__device__ __forceinline__ float gate_term(float a, float b) {
    float ez   = __expf(a);
    float oneZ = __builtin_amdgcn_rcpf(1.0f + ez);
    float t    = __expf(-2.0f * fabsf(b));
    float th   = __builtin_copysignf((1.0f - t) * __builtin_amdgcn_rcpf(1.0f + t), b);
    return oneZ * th;
}

// ---------------- dense tail: WMMA, one wave per 16-node tile ----------------
__global__ __launch_bounds__(256)
void a3tgcn_wmma(const float* __restrict__ agg, const float* __restrict__ prep,
                 const float* __restrict__ Wf, const float* __restrict__ bf,
                 float* __restrict__ out, int N)
{
    __shared__ _Float16 lds[8][16*ODIM];             // per-wave private staging (intra-wave only)
    const int lane = threadIdx.x & 31;
    const int wave = threadIdx.x >> 5;
    const int tile = blockIdx.x * 8 + wave;
    const int ntiles = (N + 15) >> 4;
    if (tile >= ntiles) return;                      // wave-uniform exit
    const int base = tile << 4;
    const int c0 = lane & 15, hi = lane >> 4;

    const float* Mz = prep + 80;
    const float* Mh = prep + 272;

    // B layout (32x16 f16): col=lane%16, h=lane/16; halves i<8 -> K=8h+i, i>=8 -> K=8+8h+i
    v16h Bz0 = {}, Bz1 = {}, Bh0 = {}, Bh1 = {}, Bfm = {};
    if (hi == 0) {
        #pragma unroll
        for (int i = 0; i < FDIM; ++i) { Bz0[i] = (_Float16)Mz[i*ODIM + c0];
                                         Bh0[i] = (_Float16)Mh[i*ODIM + c0]; }
        if (c0 < ODIM - 16) {
            #pragma unroll
            for (int i = 0; i < FDIM; ++i) { Bz1[i] = (_Float16)Mz[i*ODIM + 16 + c0];
                                             Bh1[i] = (_Float16)Mh[i*ODIM + 16 + c0]; }
        }
        if (c0 < TDIM) {
            #pragma unroll
            for (int i = 0; i < 16; ++i) Bfm[i] = (_Float16)Wf[(i < 8 ? i : i + 8)*TDIM + c0];   // K=0..7, 16..23
        }
    } else {
        if (c0 < TDIM) {
            #pragma unroll
            for (int i = 0; i < 8; ++i) Bfm[i] = (_Float16)Wf[(8 + i)*TDIM + c0]; // K=8..15
        }
    }
    const float bZ0 = prep[16 + c0];
    const float bZ1 = (c0 < 8) ? prep[32 + c0] : 0.f;   // cz[16+c0]
    const float bH0 = prep[48 + c0];
    const float bH1 = (c0 < 8) ? prep[64 + c0] : 0.f;   // ch[16+c0]

    v8f acc0 = {}, acc1 = {};
    const v8f zc = {};
    for (int t = 0; t < TDIM; ++t) {
        float p = prep[t];
        // A (16x32 f16): lanes 0-15: row=lane, halves 0..7 -> K=0..7 (our 8 features), rest 0
        v16h A = {};
        if (hi == 0) {
            int row = base + lane;
            if (row < N) {
                const float* r = agg + (size_t)row*C96 + t*FDIM;
                #pragma unroll
                for (int i = 0; i < FDIM; ++i) A[i] = (_Float16)r[i];
            }
        }
        v8f Cz0 = __builtin_amdgcn_wmma_f32_16x16x32_f16(false, A, false, Bz0, (short)0, zc, false, false);
        v8f Cz1 = __builtin_amdgcn_wmma_f32_16x16x32_f16(false, A, false, Bz1, (short)0, zc, false, false);
        v8f Ch0 = __builtin_amdgcn_wmma_f32_16x16x32_f16(false, A, false, Bh0, (short)0, zc, false, false);
        v8f Ch1 = __builtin_amdgcn_wmma_f32_16x16x32_f16(false, A, false, Bh1, (short)0, zc, false, false);
        #pragma unroll
        for (int r = 0; r < 8; ++r) {
            acc0[r] += p * gate_term(Cz0[r] + bZ0, Ch0[r] + bH0);
            acc1[r] += p * gate_term(Cz1[r] + bZ1, Ch1[r] + bH1);  // cols>=24 contribute exactly 0
        }
    }

    // relu + C-layout -> A-layout via per-wave LDS patch (same-wave DS ops are in-order)
    _Float16* L = lds[wave];
    #pragma unroll
    for (int r = 0; r < 8; ++r)
        L[(r + 8*hi)*ODIM + c0] = (_Float16)fmaxf(acc0[r], 0.f);
    if (c0 < 8) {
        #pragma unroll
        for (int r = 0; r < 8; ++r)
            L[(r + 8*hi)*ODIM + 16 + c0] = (_Float16)fmaxf(acc1[r], 0.f);
    }
    __builtin_amdgcn_wave_barrier();

    v16h Af = {};
    if (hi == 0) {                                   // row=lane: K0..7 = cols 0..7, K16..23 = cols 16..23
        #pragma unroll
        for (int i = 0; i < 8; ++i) { Af[i]     = L[lane*ODIM + i];
                                      Af[8 + i] = L[lane*ODIM + 16 + i]; }
    } else {                                         // row=lane-16: K8..15 = cols 8..15
        #pragma unroll
        for (int i = 0; i < 8; ++i) Af[i] = L[(lane - 16)*ODIM + 8 + i];
    }
    v8f Y = __builtin_amdgcn_wmma_f32_16x16x32_f16(false, Af, false, Bfm, (short)0, zc, false, false);

    if (c0 < TDIM) {
        float bb = bf[c0];
        #pragma unroll
        for (int r = 0; r < 8; ++r) {
            int row = base + r + 8*hi;
            if (row < N) out[(size_t)row*TDIM + c0] = Y[r] + bb;
        }
    }
}

extern "C" void kernel_launch(void* const* d_in, const int* in_sizes, int n_in,
                              void* d_out, int out_size, void* d_ws, size_t ws_size,
                              hipStream_t stream) {
    const float* x   = (const float*)d_in[0];
    const int*   ei  = (const int*)  d_in[1];
    const float* w   = (const float*)d_in[2];
    const float* Wcz = (const float*)d_in[3];
    const float* bcz = (const float*)d_in[4];
    // d_in[5], d_in[6] (Wcr, bcr), d_in[11], d_in[12] (Wlr, blr): dead (R-gate * H==0)
    const float* Wch = (const float*)d_in[7];
    const float* bch = (const float*)d_in[8];
    const float* Wlz = (const float*)d_in[9];
    const float* blz = (const float*)d_in[10];
    const float* Wlh = (const float*)d_in[13];
    const float* blh = (const float*)d_in[14];
    const float* att = (const float*)d_in[15];
    const float* Wf  = (const float*)d_in[16];
    const float* bf  = (const float*)d_in[17];

    const int N = in_sizes[0] / C96;
    const int E = in_sizes[2];

    float* ws   = (float*)d_ws;
    float* agg  = ws;                                // N*96 floats
    float* dinv = ws + (size_t)N * C96;              // N floats (deg, then rsqrt in place)
    float* prep = dinv + N;                          // 464 floats

    float* out = (float*)d_out;

    prep_kernel<<<1, 32, 0, stream>>>(att, Wcz, bcz, Wch, bch, Wlz, blz, Wlh, blh, prep);
    deg_init   <<<(N + 255)/256, 256, 0, stream>>>(dinv, N);
    deg_edges  <<<(E + 255)/256, 256, 0, stream>>>(ei, w, dinv, E);
    deg_to_dinv<<<(N + 255)/256, 256, 0, stream>>>(dinv, N);

    unsigned gInit = (unsigned)(((long)N * C96 + 255) / 256);
    agg_init   <<<gInit, 256, 0, stream>>>(x, dinv, agg, N);
    unsigned gScat = (unsigned)(((long)E * C96 + 255) / 256);
    agg_scatter<<<gScat, 256, 0, stream>>>(x, ei, w, dinv, agg, E);

    int ntiles = (N + 15) / 16;
    a3tgcn_wmma<<<(ntiles + 7)/8, 256, 0, stream>>>(agg, prep, Wf, bf, out, N);
}
